// LSTMCellSeq_82222853914665
// MI455X (gfx1250) — matrix operand order repaired and checked by reading, not compiled, not run on hardware
//
#include <hip/hip_runtime.h>

// ---------------------------------------------------------------------------
// LSTM scan on MI455X (gfx1250), wave32 + WMMA.
//   gates = [x_t | h_{t-1}] @ [W_ih ; W_hh]^T + (b_ih + b_hh)   (K = 512)
// Persistent kernel, 16 workgroups, device-wide barrier per timestep.
// Weights live in LDS (f16) for the whole scan; h double-buffered in f16;
// cell state c lives entirely in VGPRs (16 per lane, fixed tile ownership).
// x is pre-converted to f16 when workspace allows (removes v_cvt from the
// sequential critical path); otherwise converted in-register per fragment.
// ---------------------------------------------------------------------------

typedef __attribute__((ext_vector_type(16))) _Float16 v16h;
typedef __attribute__((ext_vector_type(8)))  _Float16 v8h;
typedef __attribute__((ext_vector_type(8)))  float    v8f;
typedef __attribute__((ext_vector_type(4)))  float    v4f;

#define T_STEPS 512
#define BATCH   256
#define DIN     256
#define HID     256
#define NWG     16
#define LDS_ROW 520                      // 512 K-halves + 8 pad (bank spread)
#define LDS_BYTES (4 * 64 * LDS_ROW * 2) // 266240 B < 320 KB WGP LDS

// ---- fragment helpers (ISA 7.12.2 layouts) --------------------------------

__device__ __forceinline__ v16h frag_from_2x8(v8h lo, v8h hi) {
  v16h r;
#pragma unroll
  for (int i = 0; i < 8; ++i) { r[i] = lo[i]; r[i + 8] = hi[i]; }
  return r;
}

// A fragment, f16 source. p = row base + lh*8 (halves). K{0-7,16-23}/{8-15,24-31}.
__device__ __forceinline__ v16h load_a_f16(const _Float16* p) {
  v8h lo = *(const v8h*)(p);
  v8h hi = *(const v8h*)(p + 16);
  return frag_from_2x8(lo, hi);
}

// A fragment, f32 source with in-register f32->f16 convert (fallback path).
__device__ __forceinline__ v16h load_a_f32(const float* p) {
  v4f a = *(const v4f*)(p);
  v4f b = *(const v4f*)(p + 4);
  v4f c = *(const v4f*)(p + 16);
  v4f d = *(const v4f*)(p + 20);
  v16h r;
#pragma unroll
  for (int i = 0; i < 4; ++i) {
    r[i]      = (_Float16)a[i];
    r[4 + i]  = (_Float16)b[i];
    r[8 + i]  = (_Float16)c[i];
    r[12 + i] = (_Float16)d[i];
  }
  return r;
}

// B fragment from LDS: column n = jcol, K{0-7,16-23}/{8-15,24-31} per lane half.
__device__ __forceinline__ v16h load_b_lds(const _Float16* lds, int g, int jcol,
                                           int k0, int lh) {
  const _Float16* p = lds + (g * 64 + jcol) * LDS_ROW + k0 + lh * 8;
  v8h lo = *(const v8h*)(p);
  v8h hi = *(const v8h*)(p + 16);
  return frag_from_2x8(lo, hi);
}

// ---- fast nonlinearities (CDNA5 has a native TANH trans op) ----------------

__device__ __forceinline__ float fast_tanh(float x) {
#if __has_builtin(__builtin_amdgcn_tanhf)
  return __builtin_amdgcn_tanhf(x);
#else
  return tanhf(x);
#endif
}
__device__ __forceinline__ float sigm(float x) {
  // sigmoid(x) = 0.5*tanh(x/2) + 0.5  -> one trans op + one FMA
  return __builtin_fmaf(0.5f, fast_tanh(0.5f * x), 0.5f);
}

// ---- device-wide barrier (16 resident WGs, monotonic counter) --------------

__device__ __forceinline__ void grid_barrier(unsigned* cnt, unsigned target) {
  __threadfence();                      // release my stores to device scope
  __syncthreads();
  if (threadIdx.x == 0) {
    __hip_atomic_fetch_add(cnt, 1u, __ATOMIC_RELEASE, __HIP_MEMORY_SCOPE_AGENT);
    while (__hip_atomic_load(cnt, __ATOMIC_ACQUIRE, __HIP_MEMORY_SCOPE_AGENT) < target) {
      __builtin_amdgcn_s_sleep(1);
    }
  }
  __syncthreads();
  __builtin_amdgcn_fence(__ATOMIC_ACQUIRE, "agent"); // invalidate stale lines
}

// ---- prep kernels ----------------------------------------------------------

__global__ void cvt_weights(const float* __restrict__ Wih, const float* __restrict__ Whh,
                            const float* __restrict__ bih, const float* __restrict__ bhh,
                            _Float16* __restrict__ Wih16, _Float16* __restrict__ Whh16,
                            float* __restrict__ bias) {
  const int tid = blockIdx.x * blockDim.x + threadIdx.x;
  const int stride = gridDim.x * blockDim.x;
  for (int i = tid; i < 4 * HID * DIN; i += stride) {
    Wih16[i] = (_Float16)Wih[i];
    Whh16[i] = (_Float16)Whh[i];
  }
  if (tid < 4 * HID) bias[tid] = bih[tid] + bhh[tid];
}

// x: f32 -> f16, vectorized 8 elements/thread/iter (b128 in, b128 out)
__global__ void cvt_x(const float* __restrict__ x, _Float16* __restrict__ x16) {
  const int n8 = (T_STEPS * BATCH * DIN) / 8;
  const int tid = blockIdx.x * blockDim.x + threadIdx.x;
  const int stride = gridDim.x * blockDim.x;
  for (int i = tid; i < n8; i += stride) {
    const v4f a = ((const v4f*)x)[2 * (size_t)i];
    const v4f b = ((const v4f*)x)[2 * (size_t)i + 1];
    v8h r;
#pragma unroll
    for (int j = 0; j < 4; ++j) {
      r[j]     = (_Float16)a[j];
      r[4 + j] = (_Float16)b[j];
    }
    ((v8h*)x16)[i] = r;
  }
}

__global__ void init_state(_Float16* __restrict__ hbuf, unsigned* __restrict__ cnt) {
  const int tid = blockIdx.x * blockDim.x + threadIdx.x;
  const int stride = gridDim.x * blockDim.x;
  for (int i = tid; i < 2 * BATCH * HID; i += stride) hbuf[i] = (_Float16)0.0f;
  if (tid == 0) *cnt = 0u;
}

// ---- persistent scan kernel ------------------------------------------------
// Grid: 16 WGs x 256 threads (8 waves). WG tile = 64 rows x 64 h-cols.
// Wave: 2 M-tiles x 4 gate-tiles (B frags reused), 8 WMMA / K-step, K=512.

template <bool X16>
__global__ void __launch_bounds__(256)
lstm_scan(const float* __restrict__ x,
          const _Float16* __restrict__ x16,
          const _Float16* __restrict__ Wih16,
          const _Float16* __restrict__ Whh16,
          const float* __restrict__ bias,
          _Float16* __restrict__ hbuf,       // 2 x (BATCH*HID) f16, double buffer
          float* __restrict__ out,           // T x BATCH x HID f32
          unsigned* __restrict__ barrier_cnt) {
  extern __shared__ _Float16 ldsW[];

  const int tid  = threadIdx.x;
  const int lane = tid & 31;
  const int wave = tid >> 5;
  const int lh   = lane >> 4;   // half-wave select (ISA A/B/C layouts)
  const int ln   = lane & 15;

  const int m0 = (blockIdx.x & 3) * 64;   // batch-row base of WG tile
  const int j0 = (blockIdx.x >> 2) * 64;  // h-column base of WG tile

  const int jloc = (wave & 3) * 16;       // N sub-tile within WG
  const int mt0  = m0 + (wave >> 2) * 32;
  const int mt1  = mt0 + 16;

  // Stage this WG's weight slice into LDS once: row r = gate*64 + local col.
  // ldsW[r][k] = W[gate*256 + j0 + jlocal][k], k: 0..255 = W_ih, 256..511 = W_hh.
  {
    const int g = tid >> 6;
    const int j = tid & 63;
    const int n = g * 256 + j0 + j;
    const v8h* s1 = (const v8h*)(Wih16 + n * DIN);
    const v8h* s2 = (const v8h*)(Whh16 + n * HID);
    v8h* d = (v8h*)(ldsW + tid * LDS_ROW);
#pragma unroll 8
    for (int i = 0; i < 32; ++i) d[i] = s1[i];
#pragma unroll 8
    for (int i = 0; i < 32; ++i) d[32 + i] = s2[i];
  }
  __syncthreads();

  const int jglob = j0 + jloc + ln;       // this lane's output column (D frag)
  const float bi = bias[jglob];
  const float bf = bias[256 + jglob];
  const float bg = bias[512 + jglob];
  const float bo = bias[768 + jglob];

  const v8f vzero = {0.f, 0.f, 0.f, 0.f, 0.f, 0.f, 0.f, 0.f};

  // Cell state lives in registers: element (p, e) -> row mt_p + lh*8 + e, col jglob.
  float creg[16];
#pragma unroll
  for (int i = 0; i < 16; ++i) creg[i] = 0.0f;

  for (int t = 0; t < T_STEPS; ++t) {
    const _Float16* hprev = hbuf + (size_t)(t & 1) * (BATCH * HID);
    _Float16* hnext       = hbuf + (size_t)((t + 1) & 1) * (BATCH * HID);

    v8f acc[2][4];
#pragma unroll
    for (int p = 0; p < 2; ++p)
#pragma unroll
      for (int g = 0; g < 4; ++g) acc[p][g] = vzero;

    // --- K = 0..255 : input x_t ---
    const float*    xr0  = x   + ((size_t)t * BATCH + (mt0 + ln)) * DIN + lh * 8;
    const float*    xr1  = x   + ((size_t)t * BATCH + (mt1 + ln)) * DIN + lh * 8;
    const _Float16* x16r0 = x16 + ((size_t)t * BATCH + (mt0 + ln)) * DIN + lh * 8;
    const _Float16* x16r1 = x16 + ((size_t)t * BATCH + (mt1 + ln)) * DIN + lh * 8;

    if (t + 1 < T_STEPS) {   // prefetch next timestep's x rows into cache
      if constexpr (X16) {
        __builtin_prefetch(x16r0 + BATCH * DIN, 0, 3);
        __builtin_prefetch(x16r1 + BATCH * DIN, 0, 3);
      } else {
        __builtin_prefetch(xr0 + BATCH * DIN, 0, 3);
        __builtin_prefetch(xr1 + BATCH * DIN, 0, 3);
      }
    }

#pragma unroll
    for (int ks = 0; ks < 8; ++ks) {
      const int k0 = ks * 32;
      v16h a0, a1;
      if constexpr (X16) {
        a0 = load_a_f16(x16r0 + k0);
        a1 = load_a_f16(x16r1 + k0);
      } else {
        a0 = load_a_f32(xr0 + k0);
        a1 = load_a_f32(xr1 + k0);
      }
#pragma unroll
      for (int g = 0; g < 4; ++g) {
        v16h b = load_b_lds(ldsW, g, jloc + ln, k0, lh);
        acc[0][g] = __builtin_amdgcn_wmma_f32_16x16x32_f16(
            false, a0, false, b, (short)0, acc[0][g], false, false);
        acc[1][g] = __builtin_amdgcn_wmma_f32_16x16x32_f16(
            false, a1, false, b, (short)0, acc[1][g], false, false);
      }
    }

    // --- K = 256..511 : recurrent h_{t-1} (f16) ---
    const _Float16* hr0 = hprev + (mt0 + ln) * HID + lh * 8;
    const _Float16* hr1 = hprev + (mt1 + ln) * HID + lh * 8;
#pragma unroll
    for (int ks = 0; ks < 8; ++ks) {
      const int k0 = ks * 32;
      v16h a0 = load_a_f16(hr0 + k0);
      v16h a1 = load_a_f16(hr1 + k0);
#pragma unroll
      for (int g = 0; g < 4; ++g) {
        v16h b = load_b_lds(ldsW, g, jloc + ln, 256 + k0, lh);
        acc[0][g] = __builtin_amdgcn_wmma_f32_16x16x32_f16(
            false, a0, false, b, (short)0, acc[0][g], false, false);
        acc[1][g] = __builtin_amdgcn_wmma_f32_16x16x32_f16(
            false, a1, false, b, (short)0, acc[1][g], false, false);
      }
    }

    // --- fused gate nonlinearity + register cell-state update + writeback ---
    float* outt = out + (size_t)t * (BATCH * HID);
#pragma unroll
    for (int p = 0; p < 2; ++p) {
      const int mt = (p == 0) ? mt0 : mt1;
#pragma unroll
      for (int e = 0; e < 8; ++e) {
        const int row = mt + lh * 8 + e;        // D-frag: M = e + 8*lh
        const int idx = row * HID + jglob;
        const float iv = acc[p][0][e] + bi;
        const float fv = acc[p][1][e] + bf;
        const float gv = acc[p][2][e] + bg;
        const float ov = acc[p][3][e] + bo;
        const float cn = sigm(fv) * creg[p * 8 + e] + sigm(iv) * fast_tanh(gv);
        const float hn = sigm(ov) * fast_tanh(cn);
        creg[p * 8 + e] = cn;
        outt[idx]  = hn;
        hnext[idx] = (_Float16)hn;
      }
    }

    if (t + 1 < T_STEPS) grid_barrier(barrier_cnt, (unsigned)(t + 1) * NWG);
  }
}

// ---------------------------------------------------------------------------

extern "C" void kernel_launch(void* const* d_in, const int* in_sizes, int n_in,
                              void* d_out, int out_size, void* d_ws, size_t ws_size,
                              hipStream_t stream) {
  const float* x   = (const float*)d_in[0];
  const float* Wih = (const float*)d_in[1];
  const float* Whh = (const float*)d_in[2];
  const float* bih = (const float*)d_in[3];
  const float* bhh = (const float*)d_in[4];
  float* out = (float*)d_out;

  // Workspace layout
  char* ws = (char*)d_ws;
  _Float16* Wih16 = (_Float16*)(ws);                 // 512 KB
  _Float16* Whh16 = (_Float16*)(ws + 524288);        // 512 KB
  float*    bias  = (float*)(ws + 1048576);          // 4 KB
  _Float16* hbuf  = (_Float16*)(ws + 1052672);       // 256 KB (2 buffers)
  unsigned* cnt   = (unsigned*)(ws + 1314816);       // 64 B slot
  _Float16* x16   = (_Float16*)(ws + 1314880);       // 64 MB (optional)
  const size_t need_x16 = 1314880 + (size_t)T_STEPS * BATCH * DIN * sizeof(_Float16);
  const bool use_x16 = ws_size >= need_x16;

  cvt_weights<<<256, 256, 0, stream>>>(Wih, Whh, bih, bhh, Wih16, Whh16, bias);
  init_state<<<64, 256, 0, stream>>>(hbuf, cnt);

  if (use_x16) {
    cvt_x<<<2048, 256, 0, stream>>>(x, x16);
    lstm_scan<true><<<NWG, 256, LDS_BYTES, stream>>>(x, x16, Wih16, Whh16, bias,
                                                     hbuf, out, cnt);
  } else {
    lstm_scan<false><<<NWG, 256, LDS_BYTES, stream>>>(x, x16, Wih16, Whh16, bias,
                                                      hbuf, out, cnt);
  }

  // h_last = out[T-1]
  hipMemcpyAsync(out + (size_t)T_STEPS * BATCH * HID,
                 out + (size_t)(T_STEPS - 1) * BATCH * HID,
                 (size_t)BATCH * HID * sizeof(float),
                 hipMemcpyDeviceToDevice, stream);
}
